// TTTLayer_83588653515431
// MI455X (gfx1250) — compile-verified, matrix operand order checked by reading, not run on hardware
//
#include <hip/hip_runtime.h>
#include <cmath>

#define B_      4
#define N_      4096
#define WIDTH_  768
#define HEADS_  12
#define HD_     64
#define MINI_   16
#define NB_     (N_ / MINI_)      // 256
#define CONV_W_ 4
#define EPS_    1e-6f
#define TTT_BASE_LR_ 1.0f

typedef __attribute__((ext_vector_type(16))) __bf16 bf16x16;
typedef __attribute__((ext_vector_type(8)))  float  f32x8;
typedef __attribute__((ext_vector_type(4)))  unsigned int v4u;
typedef __attribute__((ext_vector_type(8)))  int  v8i;
typedef __attribute__((ext_vector_type(4)))  int  v4i;

// ---------------------------------------------------------------------------
// WMMA helpers (CDNA5 wave32, V_WMMA_F32_16X16X32_BF16)
// ---------------------------------------------------------------------------
__device__ inline f32x8 wmma_bf16(bf16x16 a, bf16x16 b, f32x8 c) {
    return __builtin_amdgcn_wmma_f32_16x16x32_bf16(
        /*neg_a=*/false, a, /*neg_b=*/false, b,
        /*c_mod=*/(short)0, c, /*reuse_a=*/false, /*reuse_b=*/false);
}

__device__ inline f32x8 zero8() {
    f32x8 c;
#pragma unroll
    for (int r = 0; r < 8; ++r) c[r] = 0.0f;
    return c;
}

// K index for element-pair v (0..7) of the 16-bit A/B fragment layout.
__device__ inline int frag_kbase(int v, int hi) {
    return (v < 4) ? (hi * 8 + 2 * v) : (16 + hi * 8 + 2 * (v - 4));
}

// A-matrix fragment, row-major f32 source: element = src[m*ld + k]
__device__ inline bf16x16 frag_a_row(const float* src, int ld) {
    int lane = threadIdx.x & 31;
    int m = lane & 15, hi = lane >> 4;
    const float* r = src + m * ld;
    bf16x16 a;
#pragma unroll
    for (int v = 0; v < 8; ++v) {
        int k = frag_kbase(v, hi);
        a[2 * v]     = (__bf16)r[k];
        a[2 * v + 1] = (__bf16)r[k + 1];
    }
    return a;
}

// B-matrix fragment, B[k][n] from row-major f32 source: element = src[k*ld + n]
__device__ inline bf16x16 frag_b_col(const float* src, int ld) {
    int lane = threadIdx.x & 31;
    int n = lane & 15, hi = lane >> 4;
    bf16x16 b;
#pragma unroll
    for (int v = 0; v < 8; ++v) {
        int k = frag_kbase(v, hi);
        b[2 * v]     = (__bf16)src[k * ld + n];
        b[2 * v + 1] = (__bf16)src[(k + 1) * ld + n];
    }
    return b;
}

// K=16 (zero-padded to 32) variants
__device__ inline bf16x16 frag_a_row_k16(const float* src, int ld) {
    int lane = threadIdx.x & 31;
    int m = lane & 15, hi = lane >> 4;
    const float* r = src + m * ld;
    bf16x16 a;
#pragma unroll
    for (int v = 0; v < 8; ++v) {
        int k = frag_kbase(v, hi);
        bool ok = (v < 4);
        a[2 * v]     = ok ? (__bf16)r[k]     : (__bf16)0.0f;
        a[2 * v + 1] = ok ? (__bf16)r[k + 1] : (__bf16)0.0f;
    }
    return a;
}
__device__ inline bf16x16 frag_b_col_k16(const float* src, int ld) {
    int lane = threadIdx.x & 31;
    int n = lane & 15, hi = lane >> 4;
    bf16x16 b;
#pragma unroll
    for (int v = 0; v < 8; ++v) {
        int k = frag_kbase(v, hi);
        bool ok = (v < 4);
        b[2 * v]     = ok ? (__bf16)src[k * ld + n]       : (__bf16)0.0f;
        b[2 * v + 1] = ok ? (__bf16)src[(k + 1) * ld + n] : (__bf16)0.0f;
    }
    return b;
}

// A fragment for W1 update: A[m][k] = -(lasteta[k] * XK[k][moff+m]), K=16 padded
__device__ inline bf16x16 frag_w1upd_a(const float* XK, const float* lasteta, int moff) {
    int lane = threadIdx.x & 31;
    int m = (lane & 15) + moff, hi = lane >> 4;
    bf16x16 a;
#pragma unroll
    for (int v = 0; v < 8; ++v) {
        int k = frag_kbase(v, hi);
        float x0 = (v < 4) ? -lasteta[k] * XK[k * HD_ + m] : 0.0f;
        float x1 = (v < 4) ? -lasteta[k + 1] * XK[(k + 1) * HD_ + m] : 0.0f;
        a[2 * v]     = (__bf16)x0;
        a[2 * v + 1] = (__bf16)x1;
    }
    return a;
}

// bf16 row-major fragment from LDS (both A and B^T in the big GEMM)
__device__ inline bf16x16 frag_row_bf(const __bf16* src, int ld) {
    int lane = threadIdx.x & 31;
    int m = lane & 15, hi = lane >> 4;
    const __bf16* r = src + m * ld;
    bf16x16 a;
#pragma unroll
    for (int v = 0; v < 8; ++v) {
        int k = frag_kbase(v, hi);
        a[2 * v]     = r[k];
        a[2 * v + 1] = r[k + 1];
    }
    return a;
}

// C/D fragment (f32, 16x16): lane n = lane&15, VGPR r -> m = r + 8*(lane>>4)
__device__ inline f32x8 cfrag_load(const float* src, int ld) {
    int lane = threadIdx.x & 31;
    int n = lane & 15, hi = lane >> 4;
    f32x8 c;
#pragma unroll
    for (int r = 0; r < 8; ++r) c[r] = src[(r + 8 * hi) * ld + n];
    return c;
}
__device__ inline void cfrag_store(float* dst, int ld, f32x8 c) {
    int lane = threadIdx.x & 31;
    int n = lane & 15, hi = lane >> 4;
#pragma unroll
    for (int r = 0; r < 8; ++r) dst[(r + 8 * hi) * ld + n] = c[r];
}
__device__ inline f32x8 cfrag_bcast_row(const float* vec) {  // C[m][n] = vec[n]
    int lane = threadIdx.x & 31;
    float v = vec[lane & 15];
    f32x8 c;
#pragma unroll
    for (int r = 0; r < 8; ++r) c[r] = v;
    return c;
}

__device__ inline float wave_red_sum(float v) {
#pragma unroll
    for (int o = 16; o; o >>= 1) v += __shfl_xor(v, o, 32);
    return v;
}
// sum across an aligned 8-lane group
__device__ inline float red8(float v) {
    v += __shfl_xor(v, 1, 32);
    v += __shfl_xor(v, 2, 32);
    v += __shfl_xor(v, 4, 32);
    return v;
}

__device__ inline float gelu_tanh(float x) {
    const float k0 = 0.7978845608028654f;  // sqrt(2/pi)
    float t = tanhf(k0 * (x + 0.044715f * x * x * x));
    return 0.5f * x * (1.0f + t);
}

// ---------------------------------------------------------------------------
// TDM: DMA one 16x64 f32 tile (row stride WIDTH_ elements) from global -> LDS.
// Issued by one wave; completion via s_wait_tensorcnt.
// ---------------------------------------------------------------------------
__device__ inline void tdm_load_tile_16x64(const float* gsrc, float* lds_dst) {
    unsigned long long ga = (unsigned long long)(const void*)gsrc;
    unsigned int lds_off = (unsigned int)(unsigned long long)(void*)lds_dst; // addr[31:0] = LDS offset
    v4u g0;
    g0[0] = 1u;                                              // count=1, user mode
    g0[1] = lds_off;                                         // lds_addr
    g0[2] = (unsigned int)(ga & 0xffffffffu);                // global_addr[31:0]
    g0[3] = (unsigned int)((ga >> 32) & 0x01ffffffu) | (2u << 30); // addr[56:32] | type=2
    v8i g1;
    g1[0] = (int)(2u << 16);        // workgroup_mask=0, data_size=2 (4 bytes)
    g1[1] = (int)(64u << 16);       // tensor_dim0[15:0]=64 in bits[31:16]
    g1[2] = (int)(16u << 16);       // tensor_dim0 hi=0; tensor_dim1[15:0]=16
    g1[3] = (int)(64u << 16);       // tensor_dim1 hi=0; tile_dim0=64
    g1[4] = 16;                     // tile_dim1=16; tile_dim2=0
    g1[5] = WIDTH_;                 // tensor_dim0_stride[31:0] = 768 elements
    g1[6] = 0;                      // stride hi, tensor_dim1_stride lo
    g1[7] = 0;                      // tensor_dim1_stride hi
    v4i g2; g2[0] = 0; g2[1] = 0; g2[2] = 0; g2[3] = 0;
    v4i g3 = g2;
#if defined(__clang_major__) && (__clang_major__ >= 23)
    v8i g4; g4[0]=0; g4[1]=0; g4[2]=0; g4[3]=0; g4[4]=0; g4[5]=0; g4[6]=0; g4[7]=0;
    __builtin_amdgcn_tensor_load_to_lds(g0, g1, g2, g3, g4, 0);
#else
    __builtin_amdgcn_tensor_load_to_lds(g0, g1, g2, g3, 0);
#endif
}

// ---------------------------------------------------------------------------
// Kernel 1: tiled GEMM  C[M,N] = A[M,K] @ B[K,N]   (f32 in/out, bf16 WMMA)
// ---------------------------------------------------------------------------
#define GBM 128
#define GBN 128
#define GBK 32

__global__ __launch_bounds__(256) void gemm_f32_bf16wmma(
    const float* __restrict__ A, const float* __restrict__ B,
    float* __restrict__ C, int M, int N, int K)
{
    __shared__ __bf16 As[GBM][GBK];   // [m][k]
    __shared__ __bf16 Bs[GBN][GBK];   // [n][k]  (B transposed)

    int tid  = threadIdx.x;
    int lane = tid & 31, wid = tid >> 5;
    int wrow = wid >> 2, wcol = wid & 3;
    int mblocks = M / GBM;
    int bm = (blockIdx.x % mblocks) * GBM;
    int bn = (blockIdx.x / mblocks) * GBN;

    f32x8 acc[4][2];
#pragma unroll
    for (int mt = 0; mt < 4; ++mt)
#pragma unroll
        for (int nt = 0; nt < 2; ++nt) acc[mt][nt] = zero8();

    int arow = tid >> 1, acb = (tid & 1) * 16;
    int bkr  = tid >> 3, bcb = (tid & 7) * 16;

    for (int kb = 0; kb < K; kb += GBK) {
        {
            const float* src = A + (size_t)(bm + arow) * K + kb + acb;
#pragma unroll
            for (int j = 0; j < 16; ++j) As[arow][acb + j] = (__bf16)src[j];
        }
        {
            const float* src = B + (size_t)(kb + bkr) * N + bn + bcb;
#pragma unroll
            for (int j = 0; j < 16; ++j) Bs[bcb + j][bkr] = (__bf16)src[j];
        }
        if (kb + GBK < K) {
            __builtin_prefetch(A + (size_t)(bm + arow) * K + kb + GBK + acb, 0, 1);
            __builtin_prefetch(B + (size_t)(kb + GBK + bkr) * N + bn + bcb, 0, 1);
        }
        __syncthreads();

        bf16x16 af[4], bfr[2];
#pragma unroll
        for (int mt = 0; mt < 4; ++mt)
            af[mt] = frag_row_bf(&As[wrow * 64 + mt * 16][0], GBK);
#pragma unroll
        for (int nt = 0; nt < 2; ++nt)
            bfr[nt] = frag_row_bf(&Bs[wcol * 32 + nt * 16][0], GBK);
#pragma unroll
        for (int mt = 0; mt < 4; ++mt)
#pragma unroll
            for (int nt = 0; nt < 2; ++nt)
                acc[mt][nt] = wmma_bf16(af[mt], bfr[nt], acc[mt][nt]);
        __syncthreads();
    }

    int n = lane & 15, hi = lane >> 4;
#pragma unroll
    for (int mt = 0; mt < 4; ++mt)
#pragma unroll
        for (int nt = 0; nt < 2; ++nt)
#pragma unroll
            for (int r = 0; r < 8; ++r) {
                int row = bm + wrow * 64 + mt * 16 + r + 8 * hi;
                int col = bn + wcol * 32 + nt * 16 + n;
                C[(size_t)row * N + col] = acc[mt][nt][r];
            }
}

// ---------------------------------------------------------------------------
// Kernel 2: per-token learning-rate dots:  lrd[t][h] = dot(hs[t], lrk[h])
// ---------------------------------------------------------------------------
__global__ __launch_bounds__(128) void lrdot_kernel(
    const float* __restrict__ hs, const float* __restrict__ lrk,
    float* __restrict__ out)
{
    int t = blockIdx.x * 4 + (threadIdx.x >> 5);
    int lane = threadIdx.x & 31;
    float accv[HEADS_];
#pragma unroll
    for (int h = 0; h < HEADS_; ++h) accv[h] = 0.0f;
    for (int c = lane; c < WIDTH_; c += 32) {
        float x = hs[(size_t)t * WIDTH_ + c];
#pragma unroll
        for (int h = 0; h < HEADS_; ++h) accv[h] += x * lrk[h * WIDTH_ + c];
    }
#pragma unroll
    for (int h = 0; h < HEADS_; ++h) {
        float v = wave_red_sum(accv[h]);
        if (lane == 0) out[(size_t)t * HEADS_ + h] = v;
    }
}

// ---------------------------------------------------------------------------
// Kernel 3: causal depthwise conv (Q & K) + RoPE, one thread per channel pair
// ---------------------------------------------------------------------------
__global__ __launch_bounds__(256) void conv_rope_kernel(
    const float* __restrict__ xqk,
    const float* __restrict__ ckq, const float* __restrict__ cbq,
    const float* __restrict__ ckk, const float* __restrict__ cbk,
    float* __restrict__ XQ, float* __restrict__ XK)
{
    int idx = blockIdx.x * 256 + threadIdx.x;
    int p = idx % (WIDTH_ / 2);
    int t = idx / (WIDTH_ / 2);
    int n = t % N_;
    int b = t / N_;
    int head = p >> 5, fi = p & 31;
    int c0 = head * HD_ + 2 * fi;

    float q0 = cbq[c0], q1 = cbq[c0 + 1];
    float k0 = cbk[c0], k1 = cbk[c0 + 1];
#pragma unroll
    for (int w = 0; w < CONV_W_; ++w) {
        int tt = n - (CONV_W_ - 1) + w;
        if (tt >= 0) {
            const float* src = xqk + ((size_t)b * N_ + tt) * WIDTH_;
            float x0 = src[c0], x1 = src[c0 + 1];
            q0 += x0 * ckq[w * WIDTH_ + c0];
            q1 += x1 * ckq[w * WIDTH_ + c0 + 1];
            k0 += x0 * ckk[w * WIDTH_ + c0];
            k1 += x1 * ckk[w * WIDTH_ + c0 + 1];
        }
    }
    int pos = n & (MINI_ - 1);
    float freq = powf(10000.0f, -(float)fi / 32.0f);
    float ang = (float)pos * freq;
    float cs = cosf(ang), sn = sinf(ang);
    size_t o = (size_t)t * WIDTH_;
    XQ[o + c0]     = q0 * cs - q1 * sn;
    XQ[o + c0 + 1] = q0 * sn + q1 * cs;
    XK[o + c0]     = k0 * cs - k1 * sn;
    XK[o + c0 + 1] = k0 * sn + k1 * cs;
}

// ---------------------------------------------------------------------------
// Kernel 4: TTT scan. One workgroup (128 threads = 4 waves) per (b,h) chain.
// W1 resident in LDS; TDM stages XQ/XK/XV tiles; all matmuls bf16 WMMA.
// ---------------------------------------------------------------------------
__global__ __launch_bounds__(128) void ttt_scan_kernel(
    const float* __restrict__ XQg, const float* __restrict__ XKg,
    const float* __restrict__ XVg, const float* __restrict__ lrdot,
    const float* __restrict__ lr_bias, const float* __restrict__ tok_learn,
    const float* __restrict__ W1g, const float* __restrict__ b1g,
    const float* __restrict__ gln, const float* __restrict__ bln,
    float* __restrict__ outg)
{
    int bh = blockIdx.x;
    int b = bh / HEADS_, h = bh % HEADS_;
    int tid = threadIdx.x;
    int lane = tid & 31, wid = tid >> 5;

    __shared__ float W1[HD_ * HD_];
    __shared__ float b1s[HD_];
    __shared__ float XQ[MINI_ * HD_], XK[MINI_ * HD_], XV[MINI_ * HD_];
    __shared__ float Z1[MINI_ * HD_], grad[MINI_ * HD_], Zb[MINI_ * HD_];
    __shared__ float Attn[MINI_ * MINI_], A2[MINI_ * MINI_];
    __shared__ float lrsig[MINI_], lasteta[MINI_], tokidx[MINI_];
    __shared__ float g_s[HD_], bln_s[HD_];
    __shared__ float mu2[MINI_], rs2[MINI_];

    for (int i = tid; i < HD_ * HD_; i += 128) W1[i] = W1g[h * HD_ * HD_ + i];
    for (int i = tid; i < HD_; i += 128) {
        b1s[i]   = b1g[h * HD_ + i];
        g_s[i]   = gln[h * HD_ + i];
        bln_s[i] = bln[h * HD_ + i];
    }
    if (tid < MINI_)
        tokidx[tid] = fmaxf(1.0f / (float)(tid + 1) + tok_learn[tid], 0.0f);
    __syncthreads();

    // row-parallel reduction mapping: 16 rows x 8 lanes
    int rrow = tid >> 3, rsub = tid & 7;   // rrow 0..15, rsub 0..7 (aligned 8-lane groups)

    for (int nb = 0; nb < NB_; ++nb) {
        int t0 = nb * MINI_;
        size_t gbase = ((size_t)b * N_ + t0) * WIDTH_ + h * HD_;

        // ---- phase 0: TDM tile staging (wave 0) + learning rates (wave 1) ----
        if (wid == 0) {
            tdm_load_tile_16x64(XQg + gbase, XQ);
            tdm_load_tile_16x64(XKg + gbase, XK);
            tdm_load_tile_16x64(XVg + gbase, XV);
            __builtin_amdgcn_s_wait_tensorcnt(0);
        }
        if (tid >= 32 && tid < 32 + MINI_) {
            int j = tid - 32;
            float x = lrdot[((size_t)b * N_ + t0 + j) * HEADS_ + h] + lr_bias[h];
            float s = 1.0f / (1.0f + expf(-x));
            lrsig[j]   = s * (TTT_BASE_LR_ / (float)HD_);
            lasteta[j] = tokidx[MINI_ - 1] * lrsig[j];
        }
        __syncthreads();

        // ---- phase 1: Z1 = XK @ W1 + b1 (wave per 16-col tile); Attn (wave 0)
        {
            int nt = wid;
            f32x8 acc = cfrag_bcast_row(&b1s[nt * 16]);
            acc = wmma_bf16(frag_a_row(XK, HD_),      frag_b_col(&W1[nt * 16], HD_), acc);
            acc = wmma_bf16(frag_a_row(XK + 32, HD_), frag_b_col(&W1[32 * HD_ + nt * 16], HD_), acc);
            cfrag_store(&Z1[nt * 16], HD_, acc);
        }
        if (wid == 0) {  // Attn = tril(XQ @ XK^T)
            f32x8 acc = zero8();
            acc = wmma_bf16(frag_a_row(XQ, HD_),      frag_a_row(XK, HD_), acc);
            acc = wmma_bf16(frag_a_row(XQ + 32, HD_), frag_a_row(XK + 32, HD_), acc);
            int n = lane & 15, hi = lane >> 4;
#pragma unroll
            for (int r = 0; r < 8; ++r) {
                int m = r + 8 * hi;
                Attn[m * 16 + n] = (n <= m) ? acc[r] : 0.0f;
            }
        }
        __syncthreads();

        // ---- phase 2: grad = ln_fused_l2_bwd(Z1, XV-XK), 8 lanes per row ----
        {
            const float* zr = &Z1[rrow * HD_];
            float s = 0.0f, sq = 0.0f;
            float zv[8];
#pragma unroll
            for (int j = 0; j < 8; ++j) {
                zv[j] = zr[rsub * 8 + j];
                s += zv[j];
                sq += zv[j] * zv[j];
            }
            s = red8(s); sq = red8(sq);
            float mu   = s * (1.0f / HD_);
            float var  = sq * (1.0f / HD_) - mu * mu;
            float std  = sqrtf(var + EPS_);
            float rstd = 1.0f / std;
            float s1 = 0.0f, s2 = 0.0f;
            float xh[8], gv[8];
#pragma unroll
            for (int j = 0; j < 8; ++j) {
                int c = rsub * 8 + j;
                xh[j] = (zv[j] - mu) * rstd;
                float tgt = XV[rrow * HD_ + c] - XK[rrow * HD_ + c];
                gv[j] = (g_s[c] * xh[j] + bln_s[c] - tgt) * g_s[c];
                s1 += gv[j];
                s2 += gv[j] * xh[j];
            }
            s1 = red8(s1); s2 = red8(s2);
            float inv = rstd * (1.0f / HD_);
#pragma unroll
            for (int j = 0; j < 8; ++j) {
                int c = rsub * 8 + j;
                grad[rrow * HD_ + c] = ((float)HD_ * gv[j] - s1 - xh[j] * s2) * inv;
            }
        }
        // A2[m][k] = -(k<=m ? eta[m][k]*(Attn[m][k]+1) : 0)
        for (int i = tid; i < MINI_ * MINI_; i += 128) {
            int m = i >> 4, k = i & 15;
            float e = tokidx[m] * lrsig[k];
            A2[i] = (k <= m) ? -(e * (Attn[i] + 1.0f)) : 0.0f;
        }
        __syncthreads();

        // ---- phase 3: Z1_bar = XQ@W1 + A2@grad + b1 ----
        {
            int nt = wid;
            f32x8 acc = cfrag_bcast_row(&b1s[nt * 16]);
            acc = wmma_bf16(frag_a_row_k16(A2, 16), frag_b_col_k16(&grad[nt * 16], HD_), acc);
            acc = wmma_bf16(frag_a_row(XQ, HD_),      frag_b_col(&W1[nt * 16], HD_), acc);
            acc = wmma_bf16(frag_a_row(XQ + 32, HD_), frag_b_col(&W1[32 * HD_ + nt * 16], HD_), acc);
            cfrag_store(&Zb[nt * 16], HD_, acc);
        }
        __syncthreads();

        // ---- phase 4: W1 -= (lasteta*XK)^T @ grad; b1 update; Zb row stats ----
        {
            int mt = wid;
            bf16x16 afr = frag_w1upd_a(XK, lasteta, mt * 16);
#pragma unroll
            for (int nt = 0; nt < 4; ++nt) {
                float* wtile = &W1[(mt * 16) * HD_ + nt * 16];
                f32x8 acc = cfrag_load(wtile, HD_);
                acc = wmma_bf16(afr, frag_b_col_k16(&grad[nt * 16], HD_), acc);
                cfrag_store(wtile, HD_, acc);
            }
        }
        {
            const float* zr = &Zb[rrow * HD_];
            float s = 0.0f, sq = 0.0f;
#pragma unroll
            for (int j = 0; j < 8; ++j) {
                float v = zr[rsub * 8 + j];
                s += v;
                sq += v * v;
            }
            s = red8(s); sq = red8(sq);
            if (rsub == 0) {
                float mu  = s * (1.0f / HD_);
                float var = sq * (1.0f / HD_) - mu * mu;
                mu2[rrow] = mu;
                rs2[rrow] = 1.0f / sqrtf(var + EPS_);
            }
        }
        if (tid < HD_) {
            float s = 0.0f;
#pragma unroll
            for (int j = 0; j < MINI_; ++j) s += lasteta[j] * grad[j * HD_ + tid];
            b1s[tid] -= s;
        }
        __syncthreads();

        // ---- phase 5: out = XQ + ln_fwd(Z1_bar) ----
        for (int i = tid; i < MINI_ * HD_; i += 128) {
            int m = i >> 6, c = i & 63;
            float xhv = (Zb[i] - mu2[m]) * rs2[m];
            float val = XQ[i] + g_s[c] * xhv + bln_s[c];
            outg[((((size_t)b * HEADS_ + h) * NB_ + nb) * MINI_ + m) * HD_ + c] = val;
        }
        __syncthreads();
    }
}

// ---------------------------------------------------------------------------
// Kernel 5: gather head-major output, post-LN over 768, GELU gate, multiply
// ---------------------------------------------------------------------------
__global__ __launch_bounds__(256) void postnorm_gate_kernel(
    const float* __restrict__ outhead, const float* __restrict__ gatepre,
    const float* __restrict__ pns, const float* __restrict__ pnb,
    float* __restrict__ tmp)
{
    int t = blockIdx.x;
    int b = t / N_, n = t % N_;
    int nb = n >> 4, m = n & 15;
    int tid = threadIdx.x;
    int lane = tid & 31, wid = tid >> 5;

    __shared__ float buf[WIDTH_];
    __shared__ float red[8];

    float lsum = 0.0f;
    for (int c = tid; c < WIDTH_; c += 256) {
        int hh = c >> 6, hd = c & 63;
        float v = outhead[((((size_t)b * HEADS_ + hh) * NB_ + nb) * MINI_ + m) * HD_ + hd];
        buf[c] = v;
        lsum += v;
    }
    float s = wave_red_sum(lsum);
    if (lane == 0) red[wid] = s;
    __syncthreads();
    float tot = 0.0f;
#pragma unroll
    for (int i = 0; i < 8; ++i) tot += red[i];
    float mu = tot / (float)WIDTH_;
    __syncthreads();

    float lsq = 0.0f;
    for (int c = tid; c < WIDTH_; c += 256) {
        float d = buf[c] - mu;
        lsq += d * d;
    }
    s = wave_red_sum(lsq);
    if (lane == 0) red[wid] = s;
    __syncthreads();
    tot = 0.0f;
#pragma unroll
    for (int i = 0; i < 8; ++i) tot += red[i];
    float rstd = 1.0f / sqrtf(tot / (float)WIDTH_ + EPS_);

    for (int c = tid; c < WIDTH_; c += 256) {
        float z  = pns[c] * ((buf[c] - mu) * rstd) + pnb[c];
        float ga = gelu_tanh(gatepre[(size_t)t * WIDTH_ + c]);
        tmp[(size_t)t * WIDTH_ + c] = ga * z;
    }
}

// ---------------------------------------------------------------------------
// Host-side launch
// ---------------------------------------------------------------------------
extern "C" void kernel_launch(void* const* d_in, const int* in_sizes, int n_in,
                              void* d_out, int out_size, void* d_ws, size_t ws_size,
                              hipStream_t stream) {
    (void)in_sizes; (void)n_in; (void)out_size; (void)ws_size;
    const float* hs  = (const float*)d_in[0];
    const float* wq  = (const float*)d_in[1];
    const float* wv  = (const float*)d_in[2];
    const float* wo  = (const float*)d_in[3];
    const float* wg  = (const float*)d_in[4];
    const float* cqk = (const float*)d_in[5];
    const float* cqb = (const float*)d_in[6];
    const float* ckk = (const float*)d_in[7];
    const float* ckb = (const float*)d_in[8];
    const float* W1  = (const float*)d_in[9];
    const float* b1  = (const float*)d_in[10];
    const float* tns = (const float*)d_in[11];
    const float* tnb = (const float*)d_in[12];
    const float* lrk = (const float*)d_in[13];
    const float* lrb = (const float*)d_in[14];
    const float* lti = (const float*)d_in[15];
    const float* pns = (const float*)d_in[16];
    const float* pnb = (const float*)d_in[17];

    const size_t BNW = (size_t)B_ * N_ * WIDTH_;
    float* ws   = (float*)d_ws;
    float* xqk  = ws;               // recycled as (gate*z) before the final GEMM
    float* XV   = ws + 1 * BNW;
    float* XQ   = ws + 2 * BNW;
    float* XK   = ws + 3 * BNW;
    float* gp   = ws + 4 * BNW;
    float* oh   = ws + 5 * BNW;
    float* lrd  = ws + 6 * BNW;     // B*N*HEADS

    const int M = B_ * N_;
    const int ggrid = (M / GBM) * (WIDTH_ / GBN);

    gemm_f32_bf16wmma<<<ggrid, 256, 0, stream>>>(hs, wq, xqk, M, WIDTH_, WIDTH_);
    gemm_f32_bf16wmma<<<ggrid, 256, 0, stream>>>(hs, wv, XV,  M, WIDTH_, WIDTH_);
    gemm_f32_bf16wmma<<<ggrid, 256, 0, stream>>>(hs, wg, gp,  M, WIDTH_, WIDTH_);
    lrdot_kernel<<<M / 4, 128, 0, stream>>>(hs, lrk, lrd);
    conv_rope_kernel<<<(B_ * N_ * (WIDTH_ / 2)) / 256, 256, 0, stream>>>(
        xqk, cqk, cqb, ckk, ckb, XQ, XK);
    ttt_scan_kernel<<<B_ * HEADS_, 128, 0, stream>>>(
        XQ, XK, XV, lrd, lrb, lti, W1, b1, tns, tnb, oh);
    postnorm_gate_kernel<<<M, 256, 0, stream>>>(oh, gp, pns, pnb, xqk);
    gemm_f32_bf16wmma<<<ggrid, 256, 0, stream>>>(xqk, wo, (float*)d_out, M, WIDTH_, WIDTH_);
}
